// SolMaskedAttention_89739046682717
// MI455X (gfx1250) — matrix operand order repaired and checked by reading, not compiled, not run on hardware
//
#include <hip/hip_runtime.h>
#include <hip/hip_bf16.h>

// ---------------------------------------------------------------------------
// Causal MHA block for MI455X (gfx1250): bf16 WMMA everywhere.
//   x[4,2048,1024] -> qkv GEMM -> flash attention (16 heads, D=64) -> fc GEMM
// ---------------------------------------------------------------------------

typedef __bf16 bf16_t;
typedef bf16_t v16bf __attribute__((ext_vector_type(16)));
typedef bf16_t v8bf  __attribute__((ext_vector_type(8)));
typedef float  v8f   __attribute__((ext_vector_type(8)));

#define N_EMBD 1024
#define N_HEAD 16
#define HEAD_D 64
#define SEQ    2048
#define BATCH  4
#define ROWS   (BATCH * SEQ)   // 8192

// ---- WMMA fragment helpers -------------------------------------------------
// A fragment (16x32) in this file's contiguous k order: lane row r = lane&15, half h = lane>>4.
//   elements 0..7  = K (h*16 .. h*16+7)      -> contiguous 16B
//   elements 8..15 = K (h*16+8 .. h*16+15)   -> contiguous 16B
static __device__ inline v16bf load_a_frag(const bf16_t* p, int half) {
  union { v16bf v; v8bf h[2]; } u;
  u.h[0] = *(const v8bf*)(p + half * 16);
  u.h[1] = *(const v8bf*)(p + half * 16 + 8);
  return u.v;
}
// B fragment (32x16), same contiguous k order as A: lane col n = lane&15, half h = lane>>4 holds
// K = h*16 .. h*16+15 contiguous (caller adds h*16 to the pointer).
static __device__ inline v16bf load_b_frag(const bf16_t* p) {
  union { v16bf v; v8bf h[2]; } u;
  u.h[0] = *(const v8bf*)(p);
  u.h[1] = *(const v8bf*)(p + 8);
  return u.v;
}
static __device__ inline v8f wmma_bf16(v16bf a, v16bf b, v8f c) {
  return __builtin_amdgcn_wmma_f32_16x16x32_bf16(
      /*neg_a=*/false, a, /*neg_b=*/false, b,
      /*c_mod=*/(short)0, c, /*reuse_a=*/false, /*reuse_b=*/false);
}

// ---- conversion kernels ----------------------------------------------------
__global__ void cvt_f32_bf16(const float* __restrict__ in, bf16_t* __restrict__ out, int n) {
  int i = blockIdx.x * blockDim.x + threadIdx.x;
  if (i < n) out[i] = (bf16_t)in[i];
}

// in[K][N] (f32) -> out[N][K] (bf16)
__global__ void cvt_transpose(const float* __restrict__ in, bf16_t* __restrict__ out,
                              int K, int N) {
  int i = blockIdx.x * blockDim.x + threadIdx.x;
  if (i < K * N) {
    int k = i / N, n = i % N;
    out[(size_t)n * K + k] = (bf16_t)in[i];
  }
}

// ---- QKV projection: [8192,1024] x [1024,3072] + bias ----------------------
// One wave computes a 64x64 output tile. The 64-col tile maps exactly onto one
// head of one of {q,k,v}; epilogue scatters bf16 into attention layouts:
//   q,k: [b*16+h][t][d]   vT: [b*16+h][d][t]
__global__ __launch_bounds__(128) void gemm_qkv(
    const bf16_t* __restrict__ xb, const bf16_t* __restrict__ wT,
    const float* __restrict__ bias,
    bf16_t* __restrict__ qws, bf16_t* __restrict__ kws, bf16_t* __restrict__ vTws) {
  const int K = N_EMBD;
  int wid  = (blockIdx.x * blockDim.x + threadIdx.x) >> 5;
  int lane = threadIdx.x & 31;
  int half = lane >> 4, lr = lane & 15;
  int mt = wid & 127;   // 128 m-tiles of 64 rows
  int nt = wid >> 7;    // 48  n-tiles of 64 cols

  v8f acc[4][4];
#pragma unroll
  for (int mi = 0; mi < 4; ++mi)
#pragma unroll
    for (int ni = 0; ni < 4; ++ni) acc[mi][ni] = {};

  const bf16_t* arow[4];
  const bf16_t* brow[4];
#pragma unroll
  for (int mi = 0; mi < 4; ++mi) arow[mi] = xb + (size_t)(mt * 64 + mi * 16 + lr) * K;
#pragma unroll
  for (int ni = 0; ni < 4; ++ni) brow[ni] = wT + (size_t)(nt * 64 + ni * 16 + lr) * K;

  for (int kk = 0; kk < K; kk += 32) {
    if (kk + 32 < K) {
      __builtin_prefetch(arow[0] + kk + 32, 0, 3);   // locality 3 -> near (WGP) scope
      __builtin_prefetch(brow[0] + kk + 32, 0, 3);
    }
    v16bf a[4], b[4];
#pragma unroll
    for (int mi = 0; mi < 4; ++mi) a[mi] = load_a_frag(arow[mi] + kk, half);
#pragma unroll
    for (int ni = 0; ni < 4; ++ni) b[ni] = load_b_frag(brow[ni] + kk + half * 16);
#pragma unroll
    for (int mi = 0; mi < 4; ++mi)
#pragma unroll
      for (int ni = 0; ni < 4; ++ni) acc[mi][ni] = wmma_bf16(a[mi], b[ni], acc[mi][ni]);
  }

  int which = nt >> 4;   // 0=q 1=k 2=v
  int h     = nt & 15;
#pragma unroll
  for (int ni = 0; ni < 4; ++ni) {
    int d = ni * 16 + lr;
    float bv = bias[which * N_EMBD + h * HEAD_D + d];
#pragma unroll
    for (int mi = 0; mi < 4; ++mi) {
#pragma unroll
      for (int i = 0; i < 8; ++i) {
        int m = mt * 64 + mi * 16 + i + half * 8;
        int b_ = m >> 11, t = m & (SEQ - 1);
        bf16_t o = (bf16_t)(acc[mi][ni][i] + bv);
        size_t bh = (size_t)(b_ * N_HEAD + h);
        if (which == 0)      qws[(bh * SEQ + t) * HEAD_D + d] = o;
        else if (which == 1) kws[(bh * SEQ + t) * HEAD_D + d] = o;
        else                 vTws[(bh * HEAD_D + d) * SEQ + t] = o;
      }
    }
  }
}

// ---- flash attention: one wave per (b,h, 32-row query pair) ----------------
// Two 16-row query subtiles share every K/V fragment load: 16 WMMAs per
// 32-key strip against 8 fragment loads (halves K/V re-read traffic).
__global__ __launch_bounds__(32) void attn_kernel(
    const bf16_t* __restrict__ q, const bf16_t* __restrict__ k,
    const bf16_t* __restrict__ vT, bf16_t* __restrict__ y) {
  __shared__ __align__(16) bf16_t ptile[2][16 * 32];   // P staging (C->A layout)

  int lane = threadIdx.x & 31;
  int half = lane >> 4, lr = lane & 15;
  int gid = blockIdx.x;
  int qp = gid & 63;      // 32-row query pair within sequence
  int bh = gid >> 6;      // b*16 + h

  // Q A-fragments: two row subtiles x two 32-wide d chunks
  const bf16_t* qbase = q + ((size_t)bh * SEQ + qp * 32) * HEAD_D + lr * HEAD_D;
  v16bf qa[2][2];
#pragma unroll
  for (int r = 0; r < 2; ++r)
#pragma unroll
    for (int kc = 0; kc < 2; ++kc)
      qa[r][kc] = load_a_frag(qbase + r * 16 * HEAD_D + kc * 32, half);

  v8f o[2][4];
#pragma unroll
  for (int r = 0; r < 2; ++r)
#pragma unroll
    for (int dt = 0; dt < 4; ++dt) o[r][dt] = {};
  float rm[2][8], rl[2][8];
#pragma unroll
  for (int r = 0; r < 2; ++r)
#pragma unroll
    for (int i = 0; i < 8; ++i) { rm[r][i] = -3.0e38f; rl[r][i] = 0.0f; }

  const int nstrips = qp + 1;   // causal: keys s <= max t = qp*32+31
  for (int js = 0; js < nstrips; ++js) {
    int s0 = js * 32;
    // K B-fragments for the strip, shared by both row subtiles
    v16bf kb[2][2];
#pragma unroll
    for (int tile = 0; tile < 2; ++tile) {
      const bf16_t* krow = k + ((size_t)bh * SEQ + s0 + tile * 16 + lr) * HEAD_D;
      kb[tile][0] = load_b_frag(krow + half * 16);
      kb[tile][1] = load_b_frag(krow + 32 + half * 16);
    }
#pragma unroll
    for (int r = 0; r < 2; ++r) {
      v8f s[2];
#pragma unroll
      for (int tile = 0; tile < 2; ++tile) {
        s[tile] = {};
        s[tile] = wmma_bf16(qa[r][0], kb[tile][0], s[tile]);
        s[tile] = wmma_bf16(qa[r][1], kb[tile][1], s[tile]);
      }
      // scale, causal mask, online softmax (row m = i + 8*half, col n = lr)
      float alpha[8];
#pragma unroll
      for (int i = 0; i < 8; ++i) {
        int t = qp * 32 + r * 16 + i + half * 8;
        float v0 = s[0][i] * 0.125f;
        float v1 = s[1][i] * 0.125f;
        if (s0 + lr > t)      v0 = -1.0e30f;
        if (s0 + 16 + lr > t) v1 = -1.0e30f;
        float mx = fmaxf(v0, v1);
        mx = fmaxf(mx, __shfl_xor(mx, 1, 16));
        mx = fmaxf(mx, __shfl_xor(mx, 2, 16));
        mx = fmaxf(mx, __shfl_xor(mx, 4, 16));
        mx = fmaxf(mx, __shfl_xor(mx, 8, 16));
        float mnew = fmaxf(rm[r][i], mx);
        alpha[i] = __expf(rm[r][i] - mnew);
        rm[r][i] = mnew;
        float p0 = __expf(v0 - mnew);
        float p1 = __expf(v1 - mnew);
        float rs = p0 + p1;
        rs += __shfl_xor(rs, 1, 16);
        rs += __shfl_xor(rs, 2, 16);
        rs += __shfl_xor(rs, 4, 16);
        rs += __shfl_xor(rs, 8, 16);
        rl[r][i] = rl[r][i] * alpha[i] + rs;
        int row = i + half * 8;
        ptile[r][row * 32 + lr]      = (bf16_t)p0;
        ptile[r][row * 32 + 16 + lr] = (bf16_t)p1;
      }
#pragma unroll
      for (int dt = 0; dt < 4; ++dt)
#pragma unroll
        for (int i = 0; i < 8; ++i) o[r][dt][i] *= alpha[i];
    }
    __syncthreads();                       // LDS write -> cross-lane read
    v16bf pa[2];
#pragma unroll
    for (int r = 0; r < 2; ++r) pa[r] = load_a_frag(&ptile[r][0] + lr * 32, half);
#pragma unroll
    for (int dt = 0; dt < 4; ++dt) {
      const bf16_t* vrow =
          vT + ((size_t)bh * HEAD_D + dt * 16 + lr) * SEQ + s0 + half * 16;
      v16bf vb = load_b_frag(vrow);        // shared by both row subtiles
      o[0][dt] = wmma_bf16(pa[0], vb, o[0][dt]);
      o[1][dt] = wmma_bf16(pa[1], vb, o[1][dt]);
    }
    __syncthreads();                       // protect LDS before next strip
  }

  // normalize and store y as bf16 [b*T+t][h*64+d]
  int b_ = bh >> 4, h = bh & 15;
#pragma unroll
  for (int r = 0; r < 2; ++r) {
#pragma unroll
    for (int dt = 0; dt < 4; ++dt) {
#pragma unroll
      for (int i = 0; i < 8; ++i) {
        int t = qp * 32 + r * 16 + i + half * 8;
        float val = o[r][dt][i] / rl[r][i];
        y[((size_t)(b_ * SEQ + t)) * N_EMBD + h * HEAD_D + dt * 16 + lr] = (bf16_t)val;
      }
    }
  }
}

// ---- output projection: [8192,1024] x [1024,1024] + bias -> f32 out --------
__global__ __launch_bounds__(128) void gemm_fc(
    const bf16_t* __restrict__ yb, const bf16_t* __restrict__ wT,
    const float* __restrict__ bias, float* __restrict__ out) {
  const int K = N_EMBD;
  int wid  = (blockIdx.x * blockDim.x + threadIdx.x) >> 5;
  int lane = threadIdx.x & 31;
  int half = lane >> 4, lr = lane & 15;
  int mt = wid & 127;   // 128 m-tiles
  int nt = wid >> 7;    // 16  n-tiles

  v8f acc[4][4];
#pragma unroll
  for (int mi = 0; mi < 4; ++mi)
#pragma unroll
    for (int ni = 0; ni < 4; ++ni) acc[mi][ni] = {};

  const bf16_t* arow[4];
  const bf16_t* brow[4];
#pragma unroll
  for (int mi = 0; mi < 4; ++mi) arow[mi] = yb + (size_t)(mt * 64 + mi * 16 + lr) * K;
#pragma unroll
  for (int ni = 0; ni < 4; ++ni) brow[ni] = wT + (size_t)(nt * 64 + ni * 16 + lr) * K;

  for (int kk = 0; kk < K; kk += 32) {
    if (kk + 32 < K) {
      __builtin_prefetch(arow[0] + kk + 32, 0, 3);
      __builtin_prefetch(brow[0] + kk + 32, 0, 3);
    }
    v16bf a[4], b[4];
#pragma unroll
    for (int mi = 0; mi < 4; ++mi) a[mi] = load_a_frag(arow[mi] + kk, half);
#pragma unroll
    for (int ni = 0; ni < 4; ++ni) b[ni] = load_b_frag(brow[ni] + kk + half * 16);
#pragma unroll
    for (int mi = 0; mi < 4; ++mi)
#pragma unroll
      for (int ni = 0; ni < 4; ++ni) acc[mi][ni] = wmma_bf16(a[mi], b[ni], acc[mi][ni]);
  }

#pragma unroll
  for (int ni = 0; ni < 4; ++ni) {
    int n = nt * 64 + ni * 16 + lr;
    float bv = bias[n];
#pragma unroll
    for (int mi = 0; mi < 4; ++mi) {
#pragma unroll
      for (int i = 0; i < 8; ++i) {
        int m = mt * 64 + mi * 16 + i + half * 8;
        out[(size_t)m * N_EMBD + n] = acc[mi][ni][i] + bv;
      }
    }
  }
}

// ---------------------------------------------------------------------------
extern "C" void kernel_launch(void* const* d_in, const int* in_sizes, int n_in,
                              void* d_out, int out_size, void* d_ws, size_t ws_size,
                              hipStream_t stream) {
  (void)in_sizes; (void)n_in; (void)out_size; (void)ws_size;
  const float* x    = (const float*)d_in[0];
  const float* Wqkv = (const float*)d_in[1];
  const float* bqkv = (const float*)d_in[2];
  const float* Wfc  = (const float*)d_in[3];
  const float* bfc  = (const float*)d_in[4];
  float* out = (float*)d_out;

  char* ws = (char*)d_ws;
  size_t off = 0;
  auto take = [&](size_t bytes) {
    char* p = ws + off;
    off += (bytes + 255) & ~(size_t)255;
    return p;
  };
  bf16_t* xb   = (bf16_t*)take((size_t)ROWS * N_EMBD * 2);
  bf16_t* wqT  = (bf16_t*)take((size_t)3 * N_EMBD * N_EMBD * 2);
  bf16_t* wfT  = (bf16_t*)take((size_t)N_EMBD * N_EMBD * 2);
  bf16_t* qws  = (bf16_t*)take((size_t)ROWS * N_EMBD * 2);
  bf16_t* kws  = (bf16_t*)take((size_t)ROWS * N_EMBD * 2);
  bf16_t* vTws = (bf16_t*)take((size_t)ROWS * N_EMBD * 2);
  bf16_t* yws  = (bf16_t*)take((size_t)ROWS * N_EMBD * 2);

  cvt_f32_bf16<<<(ROWS * N_EMBD) / 256, 256, 0, stream>>>(x, xb, ROWS * N_EMBD);
  cvt_transpose<<<(3 * N_EMBD * N_EMBD) / 256, 256, 0, stream>>>(Wqkv, wqT, N_EMBD, 3 * N_EMBD);
  cvt_transpose<<<(N_EMBD * N_EMBD) / 256, 256, 0, stream>>>(Wfc, wfT, N_EMBD, N_EMBD);

  // 128 m-tiles x 48 n-tiles = 6144 waves -> 1536 blocks of 4 waves
  gemm_qkv<<<1536, 128, 0, stream>>>(xb, wqT, bqkv, qws, kws, vTws);
  // one wave per (b,h,32-row query pair): 4*16*64 = 4096 blocks
  attn_kernel<<<BATCH * N_HEAD * (SEQ / 32), 32, 0, stream>>>(qws, kws, vTws, yws);
  // 128 m-tiles x 16 n-tiles = 2048 waves -> 512 blocks
  gemm_fc<<<512, 128, 0, stream>>>(yws, wfT, bfc, out);
}
